// Net_45208825757731
// MI455X (gfx1250) — compile-verified, matrix operand order checked by reading, not run on hardware
//
#include <hip/hip_runtime.h>
#include <hip/hip_bf16.h>
#include <math.h>

// ---------------- problem constants (static shapes from reference) ----------
enum : int {
  NN    = 100000,
  EE    = 1600000,
  F_IN  = 512,
  HH    = 128,
  CC    = 40,
  DEPTH = 4,
};
// level node counts: N -> 2000 -> 1000 -> 500 -> 200
static const int NS[5] = {NN, 2000, 1000, 500, 200};

typedef __attribute__((ext_vector_type(8)))  _Float16 v8h;
typedef __attribute__((ext_vector_type(16))) _Float16 v16h;
typedef __attribute__((ext_vector_type(8)))  float    v8f;

// =======================  GEMM:  C[n x Mout] = A[n x K] @ W[K x Mout] =======
// 128 threads = 4 wave32; block computes 64 rows x 32 cols; each wave owns a
// 16x32 strip (two 16x16 accumulators -> 2 WMMAs per A fragment).
// f32 global -> f16 LDS staging; per-lane fragments are 2x ds_load_b128.
__global__ __launch_bounds__(128)
void gunet_gemm_wmma(const float* __restrict__ A, const float* __restrict__ W,
                     float* __restrict__ C, int n, int K, int Mout) {
  __shared__ _Float16 As[64][56];   // row stride 112B (16B aligned, 2-way max)
  __shared__ _Float16 Ws[32][40];   // TRANSPOSED: [col][K]; stride 80B
  const int tid  = threadIdx.x;
  const int wave = tid >> 5;
  const int lane = tid & 31;
  const int half = lane >> 4;
  const int m    = lane & 15;
  const int row0 = blockIdx.x * 64;
  const int col0 = blockIdx.y * 32;

  v8f acc0 = {0.f, 0.f, 0.f, 0.f, 0.f, 0.f, 0.f, 0.f};
  v8f acc1 = {0.f, 0.f, 0.f, 0.f, 0.f, 0.f, 0.f, 0.f};

  // staging roles (computed once; all 32-bit arithmetic)
  const int stageR = tid >> 1;
  const int stageC = (tid & 1) * 16;
  int gr = row0 + stageR; if (gr >= n) gr = n - 1;     // clamp keeps EXEC full
  const float* aSrc = A + gr * K + stageC;
  const int wCol = tid & 31;
  const int wK   = (tid >> 5) * 8;
  int gc = col0 + wCol; if (gc >= Mout) gc = Mout - 1;
  const int ar = wave * 16 + m;

  for (int kb = 0; kb < K; kb += 32) {
    // ---- stage A tile (coalesced b128 loads, packed b128 LDS stores) ----
    {
      const float4* s4 = (const float4*)(aSrc + kb);
      float4 v0 = s4[0], v1 = s4[1], v2 = s4[2], v3 = s4[3];
      v8h p0, p1;
      p0[0]=(_Float16)v0.x; p0[1]=(_Float16)v0.y; p0[2]=(_Float16)v0.z; p0[3]=(_Float16)v0.w;
      p0[4]=(_Float16)v1.x; p0[5]=(_Float16)v1.y; p0[6]=(_Float16)v1.z; p0[7]=(_Float16)v1.w;
      p1[0]=(_Float16)v2.x; p1[1]=(_Float16)v2.y; p1[2]=(_Float16)v2.z; p1[3]=(_Float16)v2.w;
      p1[4]=(_Float16)v3.x; p1[5]=(_Float16)v3.y; p1[6]=(_Float16)v3.z; p1[7]=(_Float16)v3.w;
      v8h* d = (v8h*)&As[stageR][stageC];
      d[0] = p0; d[1] = p1;
    }
    // ---- stage W tile transposed: thread -> (col, 8 consecutive K) ----
    {
      const float* ws = W + (kb + wK) * Mout + gc;
      v8h q;
#pragma unroll
      for (int i = 0; i < 8; ++i) q[i] = (_Float16)ws[i * Mout];
      *(v8h*)&Ws[wCol][wK] = q;
    }
    __syncthreads();

    // ---- fragments: element j of v16h has K = ((j>>3)<<4) + half*8 + (j&7),
    //      i.e. two contiguous 8-half runs -> two b128 LDS loads each ----
    const v8h* arow = (const v8h*)&As[ar][0];
    v8h a0 = arow[half], a1 = arow[2 + half];
    v16h av = __builtin_shufflevector(a0, a1, 0,1,2,3,4,5,6,7,8,9,10,11,12,13,14,15);

    const v8h* bp0 = (const v8h*)&Ws[m][0];
    v8h b00 = bp0[half], b01 = bp0[2 + half];
    v16h bv0 = __builtin_shufflevector(b00, b01, 0,1,2,3,4,5,6,7,8,9,10,11,12,13,14,15);

    const v8h* bp1 = (const v8h*)&Ws[m + 16][0];
    v8h b10 = bp1[half], b11 = bp1[2 + half];
    v16h bv1 = __builtin_shufflevector(b10, b11, 0,1,2,3,4,5,6,7,8,9,10,11,12,13,14,15);

    acc0 = __builtin_amdgcn_wmma_f32_16x16x32_f16(false, av, false, bv0,
                                                  (short)0, acc0, false, false);
    acc1 = __builtin_amdgcn_wmma_f32_16x16x32_f16(false, av, false, bv1,
                                                  (short)0, acc1, false, false);
    __syncthreads();
  }

  // ---- store: lane -> N=m; VGPR v -> M = v + half*8 ----
  const int colg0 = col0 + m;
  const int colg1 = colg0 + 16;
  const int rbase = row0 + wave * 16 + half * 8;
  if ((row0 + 64 <= n) && (col0 + 32 <= Mout)) {     // interior fast path
    float* cp = C + rbase * Mout;
#pragma unroll
    for (int v = 0; v < 8; ++v, cp += Mout) {
      cp[colg0] = acc0[v];
      cp[colg1] = acc1[v];
    }
  } else {                                           // edge blocks
    int base = rbase * Mout;
#pragma unroll
    for (int v = 0; v < 8; ++v, base += Mout) {
      if (rbase + v < n) {
        if (colg0 < Mout) C[base + colg0] = acc0[v];
        if (colg1 < Mout) C[base + colg1] = acc1[v];
      }
    }
  }
}

// ======================= GCN normalization ==================================
__global__ void gunet_deg_init(float* deg, int n) {
  int i = blockIdx.x * blockDim.x + threadIdx.x;
  if (i < n) deg[i] = 2.0f;                       // improved=True self-loop
}
__global__ void gunet_deg_acc(const int* __restrict__ col,
                              const float* __restrict__ ew, int E, float* deg) {
  int e = blockIdx.x * blockDim.x + threadIdx.x;
  if (e >= E) return;
  float w = ew ? ew[e] : 1.0f;
  if (w != 0.0f) atomicAdd(&deg[col[e]], w);
}
__global__ void gunet_dis(const float* deg, float* dis, int n) {
  int i = blockIdx.x * blockDim.x + threadIdx.x;
  if (i >= n) return;
  float d = deg[i];
  dis[i] = (d > 0.0f) ? rsqrtf(fmaxf(d, 1e-12f)) : 0.0f;
}

// out[i][f] = lin[i][f] * 2*dis[i]^2   (self-loop term; also initializes out)
template <int F>
__global__ void gunet_selfloop(const float* __restrict__ lin,
                               const float* __restrict__ dis,
                               float* __restrict__ out, int n) {
  int idx = blockIdx.x * blockDim.x + threadIdx.x;
  if (idx >= n * F) return;
  float d = dis[idx / F];
  out[idx] = lin[idx] * (2.0f * d * d);
}

// out[c] += lin[r] * (dis[r]*w*dis[c]); one wave32 per edge
template <int F>
__global__ void gunet_edge_scatter(const int* __restrict__ row,
                                   const int* __restrict__ col,
                                   const float* __restrict__ ew, int E,
                                   const float* __restrict__ lin,
                                   const float* __restrict__ dis,
                                   float* __restrict__ out) {
  int e    = blockIdx.x * (blockDim.x >> 5) + (threadIdx.x >> 5);
  int lane = threadIdx.x & 31;
  if (e >= E) return;
  float w = ew ? ew[e] : 1.0f;
  if (w == 0.0f) return;                        // pooled-out edges
  int r = row[e], c = col[e];
  float nrm = dis[r] * w * dis[c];
  const float* src = lin + r * F;
  float*       dst = out + c * F;
  __builtin_prefetch(src, 0, 1);                // global_prefetch_b8
  if (F == 128) {                               // 512B row: one float4/lane
    float4 v = *(const float4*)(src + lane * 4);
    atomicAdd(dst + lane * 4 + 0, v.x * nrm);
    atomicAdd(dst + lane * 4 + 1, v.y * nrm);
    atomicAdd(dst + lane * 4 + 2, v.z * nrm);
    atomicAdd(dst + lane * 4 + 3, v.w * nrm);
  } else {
#pragma unroll
    for (int f = lane; f < F; f += 32) atomicAdd(dst + f, src[f] * nrm);
  }
}

template <int F>
__global__ void gunet_bias_act(float* out, const float* __restrict__ b,
                               int n, int relu) {
  int idx = blockIdx.x * blockDim.x + threadIdx.x;
  if (idx >= n * F) return;
  float v = out[idx] + b[idx % F];
  out[idx] = relu ? fmaxf(v, 0.0f) : v;
}

// ======================= TopK pooling =======================================
__global__ void gunet_attn_norm(const float* attn, float* out, int H) {
  __shared__ float red[128];
  int t = threadIdx.x;
  float s = 0.f;
  for (int i = t; i < H; i += blockDim.x) s += attn[i] * attn[i];
  red[t] = s; __syncthreads();
  for (int off = 64; off > 0; off >>= 1) {
    if (t < off) red[t] += red[t + off];
    __syncthreads();
  }
  if (t == 0) out[0] = sqrtf(red[0]);
}

// score[i] = tanh(h[i].attn / ||attn||); order-preserving uint key (H=128)
__global__ void gunet_score(const float* __restrict__ h,
                            const float* __restrict__ attn,
                            const float* __restrict__ normp,
                            float* score, unsigned* keys, int n) {
  int i    = blockIdx.x * (blockDim.x >> 5) + (threadIdx.x >> 5);
  int lane = threadIdx.x & 31;
  if (i >= n) return;
  float4 hv = *(const float4*)(h + i * HH + lane * 4);
  float4 av = *(const float4*)(attn + lane * 4);
  float s = hv.x * av.x + hv.y * av.y + hv.z * av.z + hv.w * av.w;
  for (int off = 16; off > 0; off >>= 1) s += __shfl_xor(s, off, 32);
  if (lane == 0) {
    float sc = tanhf(s / normp[0]);
    score[i] = sc;
    unsigned u = __float_as_uint(sc);
    keys[i] = (u & 0x80000000u) ? ~u : (u | 0x80000000u);
  }
}

__global__ void gunet_topk_init(unsigned* state, int* cnt, int* cntG,
                                int* ctrA, int* ctrB) {
  state[0] = 0u; *cnt = 0; *cntG = 0; *ctrA = 0; *ctrB = 0;
}
// count keys >= (prefix | 1<<bit)
__global__ void gunet_count_ge(const unsigned* __restrict__ keys, int n,
                               const unsigned* __restrict__ state, int bit,
                               int* cnt) {
  __shared__ int sc;
  if (threadIdx.x == 0) sc = 0;
  __syncthreads();
  unsigned cand = state[0] | (1u << bit);
  int i = blockIdx.x * blockDim.x + threadIdx.x;
  int v = (i < n && keys[i] >= cand) ? 1 : 0;
  for (int off = 16; off > 0; off >>= 1) v += __shfl_xor(v, off, 32);
  if ((threadIdx.x & 31) == 0 && v) atomicAdd(&sc, v);
  __syncthreads();
  if (threadIdx.x == 0 && sc) atomicAdd(cnt, sc);
}
__global__ void gunet_select_bit(unsigned* state, int* cnt, int k, int bit) {
  if (*cnt >= k) state[0] |= (1u << bit);
  *cnt = 0;
}
__global__ void gunet_count_gt(const unsigned* __restrict__ keys, int n,
                               const unsigned* __restrict__ state, int* cntG) {
  __shared__ int sc;
  if (threadIdx.x == 0) sc = 0;
  __syncthreads();
  unsigned p = state[0];
  int i = blockIdx.x * blockDim.x + threadIdx.x;
  int v = (i < n && keys[i] > p) ? 1 : 0;
  for (int off = 16; off > 0; off >>= 1) v += __shfl_xor(v, off, 32);
  if ((threadIdx.x & 31) == 0 && v) atomicAdd(&sc, v);
  __syncthreads();
  if (threadIdx.x == 0 && sc) atomicAdd(cntG, sc);
}
__global__ void gunet_compact_gt(const unsigned* keys, int n,
                                 const unsigned* state, int* ctr, int* perm) {
  int i = blockIdx.x * blockDim.x + threadIdx.x;
  if (i < n && keys[i] > state[0]) perm[atomicAdd(ctr, 1)] = i;
}
__global__ void gunet_compact_eq(const unsigned* keys, int n,
                                 const unsigned* state, const int* cntG,
                                 int* ctr, int* perm, int k) {
  int i = blockIdx.x * blockDim.x + threadIdx.x;
  if (i < n && keys[i] == state[0]) {
    int p = cntG[0] + atomicAdd(ctr, 1);
    if (p < k) perm[p] = i;
  }
}

__global__ void gunet_gate(const float* __restrict__ h,
                           const float* __restrict__ score,
                           const int* __restrict__ perm,
                           float* __restrict__ hp, int k) {
  int idx = blockIdx.x * blockDim.x + threadIdx.x;
  if (idx >= k * HH) return;
  int s = perm[idx >> 7];
  hp[idx] = h[s * HH + (idx & 127)] * score[s];
}

__global__ void gunet_fill_i32(int* p, int v, int n) {
  int i = blockIdx.x * blockDim.x + threadIdx.x;
  if (i < n) p[i] = v;
}
__global__ void gunet_set_newidx(const int* perm, int* newidx, int k) {
  int j = blockIdx.x * blockDim.x + threadIdx.x;
  if (j < k) newidx[perm[j]] = j;
}
__global__ void gunet_relabel(const int* __restrict__ rI, const int* __restrict__ cI,
                              const float* __restrict__ eI, int E,
                              const int* __restrict__ newidx,
                              int* rO, int* cO, float* eO) {
  int e = blockIdx.x * blockDim.x + threadIdx.x;
  if (e >= E) return;
  int r2 = newidx[rI[e]], c2 = newidx[cI[e]];
  bool valid = (r2 >= 0) && (c2 >= 0);
  float w = eI ? eI[e] : 1.0f;
  rO[e] = valid ? r2 : 0;
  cO[e] = valid ? c2 : 0;
  eO[e] = valid ? w : 0.0f;
}

// dst[perm[j]] += src[j]   (unpool; perm entries unique)
__global__ void gunet_scatter_add_rows(float* __restrict__ dst,
                                       const int* __restrict__ perm,
                                       const float* __restrict__ src, int k) {
  int idx = blockIdx.x * blockDim.x + threadIdx.x;
  if (idx >= k * HH) return;
  dst[perm[idx >> 7] * HH + (idx & 127)] += src[idx];
}

// ======================= log_softmax ========================================
__global__ void gunet_log_softmax(const float* __restrict__ in,
                                  float* __restrict__ out, int n) {
  int i    = blockIdx.x * (blockDim.x >> 5) + (threadIdx.x >> 5);
  int lane = threadIdx.x & 31;
  if (i >= n) return;
  const float* r = in + i * CC;
  float mx = -INFINITY;
#pragma unroll
  for (int c = lane; c < CC; c += 32) mx = fmaxf(mx, r[c]);
  for (int off = 16; off > 0; off >>= 1) mx = fmaxf(mx, __shfl_xor(mx, off, 32));
  float s = 0.f;
#pragma unroll
  for (int c = lane; c < CC; c += 32) s += expf(r[c] - mx);
  for (int off = 16; off > 0; off >>= 1) s += __shfl_xor(s, off, 32);
  float lse = mx + logf(s);
#pragma unroll
  for (int c = lane; c < CC; c += 32) out[i * CC + c] = r[c] - lse;
}

// ======================= host-side helpers ==================================
static inline int g1(long n, int b = 256) { return (int)((n + b - 1) / b); }

static void run_gemm(const float* A, const float* W, float* C,
                     int n, int K, int Mout, hipStream_t s) {
  dim3 grid((n + 63) / 64, (Mout + 31) / 32);
  gunet_gemm_wmma<<<grid, 128, 0, s>>>(A, W, C, n, K, Mout);
}
static void run_dis(const int* col, const float* ew, int E, int n,
                    float* deg, float* dis, hipStream_t s) {
  gunet_deg_init<<<g1(n), 256, 0, s>>>(deg, n);
  gunet_deg_acc<<<g1(E), 256, 0, s>>>(col, ew, E, deg);
  gunet_dis<<<g1(n), 256, 0, s>>>(deg, dis, n);
}
static void run_gcn(const float* h, const int* row, const int* col,
                    const float* ew, const float* W, const float* b,
                    const float* dis, float* lin, float* outb,
                    int n, int E, int Fin, int Fout, int relu, hipStream_t s) {
  run_gemm(h, W, lin, n, Fin, Fout, s);
  if (Fout == HH) {
    gunet_selfloop<HH><<<g1((long)n * HH), 256, 0, s>>>(lin, dis, outb, n);
    gunet_edge_scatter<HH><<<(E + 7) / 8, 256, 0, s>>>(row, col, ew, E, lin, dis, outb);
    gunet_bias_act<HH><<<g1((long)n * HH), 256, 0, s>>>(outb, b, n, relu);
  } else {
    gunet_selfloop<CC><<<g1((long)n * CC), 256, 0, s>>>(lin, dis, outb, n);
    gunet_edge_scatter<CC><<<(E + 7) / 8, 256, 0, s>>>(row, col, ew, E, lin, dis, outb);
    gunet_bias_act<CC><<<g1((long)n * CC), 256, 0, s>>>(outb, b, n, relu);
  }
}

// ======================= entry point ========================================
extern "C" void kernel_launch(void* const* d_in, const int* in_sizes, int n_in,
                              void* d_out, int out_size, void* d_ws, size_t ws_size,
                              hipStream_t stream) {
  (void)in_sizes; (void)n_in; (void)out_size; (void)ws_size;
  const float* x    = (const float*)d_in[0];
  const int*   ei   = (const int*)  d_in[1];
  const int*   row0 = ei;
  const int*   col0 = ei + EE;
  const float* w0   = (const float*)d_in[2];
  const float* b0   = (const float*)d_in[3];
  const float* dw   = (const float*)d_in[4];   // [4][H][H]
  const float* db   = (const float*)d_in[5];   // [4][H]
  const float* pw   = (const float*)d_in[6];   // [4][H]
  const float* uw   = (const float*)d_in[7];   // [3][H][H]
  const float* ub   = (const float*)d_in[8];   // [3][H]
  const float* wout = (const float*)d_in[9];   // [H][C]
  const float* bout = (const float*)d_in[10];  // [C]
  float* out = (float*)d_out;

  // -------- workspace carve-out (all state lives here; rewritten each call)
  char* p = (char*)d_ws;
  auto alloc = [&](size_t bytes) -> void* {
    void* q = (void*)p;
    p += (bytes + 255) & ~(size_t)255;
    return q;
  };
  float* lin  = (float*)alloc((size_t)NN * HH * 4);
  float* xs0  = (float*)alloc((size_t)NN * HH * 4);
  float* gout = (float*)alloc((size_t)NN * CC * 4);
  float* hA   = (float*)alloc((size_t)NS[1] * HH * 4);
  float* hB   = (float*)alloc((size_t)NS[1] * HH * 4);
  float* xsL[4];
  xsL[0] = xs0;
  for (int i = 1; i < 4; ++i) xsL[i] = (float*)alloc((size_t)NS[i] * HH * 4);
  int*   grow[4]; int* gcol[4]; float* gew[4];
  for (int i = 0; i < 4; ++i) {
    grow[i] = (int*)  alloc((size_t)EE * 4);
    gcol[i] = (int*)  alloc((size_t)EE * 4);
    gew[i]  = (float*)alloc((size_t)EE * 4);
  }
  float* deg = (float*)alloc((size_t)NN * 4);
  float* disL[5];
  for (int l = 0; l < 5; ++l) disL[l] = (float*)alloc((size_t)NS[l] * 4);
  float*    score  = (float*)   alloc((size_t)NN * 4);
  unsigned* keys   = (unsigned*)alloc((size_t)NN * 4);
  int*      newidx = (int*)     alloc((size_t)NN * 4);
  int* perm[4];
  for (int i = 0; i < 4; ++i) perm[i] = (int*)alloc((size_t)NS[i + 1] * 4);
  int* tk = (int*)alloc(8 * 4);
  unsigned* tstate = (unsigned*)&tk[0];
  int *tcnt = &tk[1], *tcntG = &tk[2], *tctrA = &tk[3], *tctrB = &tk[4];

  auto run_topk = [&](const float* h, const float* attn, int n, int k,
                      int* permO) {
    gunet_attn_norm<<<1, 128, 0, stream>>>(attn, (float*)&tk[5], HH);
    gunet_score<<<(n + 7) / 8, 256, 0, stream>>>(h, attn, (float*)&tk[5],
                                                 score, keys, n);
    gunet_topk_init<<<1, 1, 0, stream>>>(tstate, tcnt, tcntG, tctrA, tctrB);
    for (int bit = 31; bit >= 0; --bit) {
      gunet_count_ge<<<g1(n), 256, 0, stream>>>(keys, n, tstate, bit, tcnt);
      gunet_select_bit<<<1, 1, 0, stream>>>(tstate, tcnt, k, bit);
    }
    gunet_count_gt<<<g1(n), 256, 0, stream>>>(keys, n, tstate, tcntG);
    gunet_compact_gt<<<g1(n), 256, 0, stream>>>(keys, n, tstate, tctrA, permO);
    gunet_compact_eq<<<g1(n), 256, 0, stream>>>(keys, n, tstate, tcntG, tctrB,
                                                permO, k);
  };

  // =================== down path ===========================================
  run_dis(col0, nullptr, EE, NN, deg, disL[0], stream);
  run_gcn(x, row0, col0, nullptr, w0, b0, disL[0], lin, xs0,
          NN, EE, F_IN, HH, /*relu=*/1, stream);

  const int*   curRow = row0;
  const int*   curCol = col0;
  const float* curEw  = nullptr;
  const float* h      = xs0;
  int ncur = NN;

  for (int i = 0; i < DEPTH; ++i) {
    int k = NS[i + 1];
    run_topk(h, pw + (size_t)i * HH, ncur, k, perm[i]);
    gunet_gate<<<g1((long)k * HH), 256, 0, stream>>>(h, score, perm[i], hA, k);
    gunet_fill_i32<<<g1(ncur), 256, 0, stream>>>(newidx, -1, ncur);
    gunet_set_newidx<<<g1(k), 256, 0, stream>>>(perm[i], newidx, k);
    gunet_relabel<<<g1(EE), 256, 0, stream>>>(curRow, curCol, curEw, EE, newidx,
                                              grow[i], gcol[i], gew[i]);
    curRow = grow[i]; curCol = gcol[i]; curEw = gew[i]; ncur = k;
    run_dis(curCol, curEw, EE, ncur, deg, disL[i + 1], stream);
    float* outb = (i < DEPTH - 1) ? xsL[i + 1] : hB;
    run_gcn(hA, curRow, curCol, curEw, dw + (size_t)i * HH * HH,
            db + (size_t)i * HH, disL[i + 1], lin, outb,
            ncur, EE, HH, HH, /*relu=*/1, stream);
    h = outb;
  }

  // =================== up path (sum_res=True) ==============================
  const float* hup = hB;                 // 200 x H
  for (int i = 0; i < DEPTH; ++i) {
    int j = DEPTH - 1 - i;
    int nj = NS[j];
    // res (xsL[j]) += scatter(perm[j], hup)
    gunet_scatter_add_rows<<<g1((long)NS[j + 1] * HH), 256, 0, stream>>>(
        xsL[j], perm[j], hup, NS[j + 1]);
    const int*   r = (j == 0) ? row0 : grow[j - 1];
    const int*   c = (j == 0) ? col0 : gcol[j - 1];
    const float* w = (j == 0) ? nullptr : gew[j - 1];
    if (i < DEPTH - 1) {
      run_gcn(xsL[j], r, c, w, uw + (size_t)i * HH * HH, ub + (size_t)i * HH,
              disL[j], lin, hA, nj, EE, HH, HH, /*relu=*/1, stream);
      hup = hA;
    } else {
      run_gcn(xsL[j], r, c, w, wout, bout, disL[j], lin, gout,
              nj, EE, HH, CC, /*relu=*/0, stream);
    }
  }

  gunet_log_softmax<<<(NN + 7) / 8, 256, 0, stream>>>(gout, out, NN);
}
